// GeoSlotV2_19550691131582
// MI455X (gfx1250) — matrix operand order repaired and checked by reading, not compiled, not run on hardware
//
#include <hip/hip_runtime.h>
#include <math.h>

#define BB 32
#define NN 784
#define FF 1568
#define DD 256
#define KK 8
#define HH 4
#define DH 64

typedef __attribute__((ext_vector_type(16))) __bf16 v16bf;
typedef __attribute__((ext_vector_type(8)))  __bf16 v8bf;
typedef __attribute__((ext_vector_type(8)))  float  v8f;

union ABu { v16bf v16; v8bf v8[2]; };

__device__ __forceinline__ float block_reduce_sum(float v, float* red) {
  int t = threadIdx.x;
  red[t] = v; __syncthreads();
  for (int s = blockDim.x >> 1; s > 0; s >>= 1) {
    if (t < s) red[t] += red[t + s];
    __syncthreads();
  }
  float r = red[0];
  __syncthreads();
  return r;
}

// ---- transpose+convert weight [F][D] f32 -> [D][F] bf16 ----
__global__ void k_wt(const float* __restrict__ W, __bf16* __restrict__ Wt) {
  int idx = blockIdx.x * 256 + threadIdx.x;
  if (idx >= FF * DD) return;
  int f = idx / DD, d = idx % DD;
  Wt[(size_t)d * FF + f] = (__bf16)W[idx];
}

// ---- LayerNorm over F, write bf16 ----
__global__ __launch_bounds__(256) void k_ln_feat(const float* __restrict__ X,
                                                 const float* __restrict__ g,
                                                 const float* __restrict__ b,
                                                 __bf16* __restrict__ Y) {
  __shared__ float red[256];
  int row = blockIdx.x;
  const float* x = X + (size_t)row * FF;
  __bf16* y = Y + (size_t)row * FF;
  float s1 = 0.f, s2 = 0.f;
  for (int f = threadIdx.x; f < FF; f += 256) { float v = x[f]; s1 += v; s2 += v * v; }
  s1 = block_reduce_sum(s1, red);
  s2 = block_reduce_sum(s2, red);
  float m  = s1 * (1.0f / FF);
  float var = s2 * (1.0f / FF) - m * m;
  float rs = rsqrtf(var + 1e-5f);
  for (int f = threadIdx.x; f < FF; f += 256)
    y[f] = (__bf16)((x[f] - m) * rs * g[f] + b[f]);
}

// ---- Fused WMMA GEMM: OK[m][n] = sum_k X[m][k]*WtK[n][k], OV likewise.
//      X bf16 [25088][1568], WtK/WtV bf16 [256][1568]. A read once for both weights. ----
__global__ __launch_bounds__(256) void k_gemm(const __bf16* __restrict__ X,
                                              const __bf16* __restrict__ WtK,
                                              const __bf16* __restrict__ WtV,
                                              float* __restrict__ OK_,
                                              float* __restrict__ OV_) {
  int wave = threadIdx.x >> 5;
  int lane = threadIdx.x & 31;
  int mBase = blockIdx.x * 32 + (wave >> 2) * 16;
  int nBase = (wave & 3) * 64;
  int lid  = lane & 15;
  int hi   = lane >> 4;          // 0 or 1
  int m    = mBase + lid;

  v8f ck0 = {}, ck1 = {}, ck2 = {}, ck3 = {};
  v8f cv0 = {}, cv1 = {}, cv2 = {}, cv3 = {};
  const __bf16* arow = X + (size_t)m * FF;
  size_t boff = (size_t)(nBase + lid) * FF;
  const __bf16* bk0 = WtK + boff;
  const __bf16* bk1 = WtK + boff + (size_t)16 * FF;
  const __bf16* bk2 = WtK + boff + (size_t)32 * FF;
  const __bf16* bk3 = WtK + boff + (size_t)48 * FF;
  const __bf16* bv0 = WtV + boff;
  const __bf16* bv1 = WtV + boff + (size_t)16 * FF;
  const __bf16* bv2 = WtV + boff + (size_t)32 * FF;
  const __bf16* bv3 = WtV + boff + (size_t)48 * FF;

  for (int k0 = 0; k0 < FF; k0 += 32) {
    // prefetch the A stream ahead (emits global_prefetch)
    if (k0 + 512 < FF) __builtin_prefetch(arow + k0 + 512, 0, 0);
    // A fragment: lanes<16 hold K {0..15}; lanes>=16 hold K {16..31}
    ABu a;
    a.v8[0] = *(const v8bf*)(arow + k0 + hi * 16);
    a.v8[1] = *(const v8bf*)(arow + k0 + hi * 16 + 8);
    // B fragments: lane holds 16 contiguous K at offset hi*16, col n = nBase + lid (+16/32/48)
    int kb = k0 + hi * 16;
    v16bf B0 = *(const v16bf*)(bk0 + kb);
    v16bf B1 = *(const v16bf*)(bk1 + kb);
    v16bf B2 = *(const v16bf*)(bk2 + kb);
    v16bf B3 = *(const v16bf*)(bk3 + kb);
    v16bf B4 = *(const v16bf*)(bv0 + kb);
    v16bf B5 = *(const v16bf*)(bv1 + kb);
    v16bf B6 = *(const v16bf*)(bv2 + kb);
    v16bf B7 = *(const v16bf*)(bv3 + kb);
    ck0 = __builtin_amdgcn_wmma_f32_16x16x32_bf16(false, a.v16, false, B0, (short)0, ck0, false, false);
    ck1 = __builtin_amdgcn_wmma_f32_16x16x32_bf16(false, a.v16, false, B1, (short)0, ck1, false, false);
    ck2 = __builtin_amdgcn_wmma_f32_16x16x32_bf16(false, a.v16, false, B2, (short)0, ck2, false, false);
    ck3 = __builtin_amdgcn_wmma_f32_16x16x32_bf16(false, a.v16, false, B3, (short)0, ck3, false, false);
    cv0 = __builtin_amdgcn_wmma_f32_16x16x32_bf16(false, a.v16, false, B4, (short)0, cv0, false, false);
    cv1 = __builtin_amdgcn_wmma_f32_16x16x32_bf16(false, a.v16, false, B5, (short)0, cv1, false, false);
    cv2 = __builtin_amdgcn_wmma_f32_16x16x32_bf16(false, a.v16, false, B6, (short)0, cv2, false, false);
    cv3 = __builtin_amdgcn_wmma_f32_16x16x32_bf16(false, a.v16, false, B7, (short)0, cv3, false, false);
  }
  // D layout: VGPR r holds M = r (lanes 0-15) / r+8 (lanes 16-31), N = lane%16
  for (int r = 0; r < 8; r++) {
    int mm = mBase + r + hi * 8;
    float* okrow = OK_ + (size_t)mm * DD;
    float* ovrow = OV_ + (size_t)mm * DD;
    okrow[nBase + lid]      = ck0[r];
    okrow[nBase + 16 + lid] = ck1[r];
    okrow[nBase + 32 + lid] = ck2[r];
    okrow[nBase + 48 + lid] = ck3[r];
    ovrow[nBase + lid]      = cv0[r];
    ovrow[nBase + 16 + lid] = cv1[r];
    ovrow[nBase + 32 + lid] = cv2[r];
    ovrow[nBase + 48 + lid] = cv3[r];
  }
}

// ---- broadcast slots_init ----
__global__ void k_slot_init(const float* __restrict__ init, float* __restrict__ slots) {
  int idx = blockIdx.x * 256 + threadIdx.x;
  if (idx < BB * KK * DD) slots[idx] = init[idx % (KK * DD)];
}

// ---- q = ln(slots) @ W_q ----
__global__ __launch_bounds__(256) void k_qproj(const float* __restrict__ slots,
                                               const float* __restrict__ g,
                                               const float* __restrict__ b,
                                               const float* __restrict__ Wq,
                                               float* __restrict__ Q) {
  __shared__ float red[256];
  __shared__ float sl[256];
  int row = blockIdx.x, t = threadIdx.x;
  float v = slots[(size_t)row * DD + t];
  float s1 = block_reduce_sum(v, red);
  float s2 = block_reduce_sum(v * v, red);
  float m = s1 * (1.0f / DD);
  float var = s2 * (1.0f / DD) - m * m;
  float rs = rsqrtf(var + 1e-5f);
  sl[t] = (v - m) * rs * g[t] + b[t];
  __syncthreads();
  float acc = 0.f;
  for (int j = 0; j < DD; j++) acc += sl[j] * Wq[(size_t)j * DD + t];
  Q[(size_t)row * DD + t] = acc;
}

// ---- logits + softmax over slots: thread per (b,h,n) ----
__global__ __launch_bounds__(128) void k_attn(const float* __restrict__ Q,
                                              const float* __restrict__ Km,
                                              float* __restrict__ A) {
  int idx = blockIdx.x * 128 + threadIdx.x;
  if (idx >= BB * HH * NN) return;
  int n = idx % NN;
  int bh = idx / NN;
  int h = bh % HH, b = bh / HH;
  const float* krow = Km + ((size_t)(b * NN + n)) * DD + h * DH;
  const float* qbase = Q + (size_t)b * KK * DD + h * DH;
  float acc[KK];
#pragma unroll
  for (int k = 0; k < KK; k++) acc[k] = 0.f;
  for (int d = 0; d < DH; d++) {
    float kv = krow[d];
#pragma unroll
    for (int k = 0; k < KK; k++) acc[k] += kv * qbase[k * DD + d];
  }
  float mx = -1e30f;
#pragma unroll
  for (int k = 0; k < KK; k++) { acc[k] *= 0.125f; mx = fmaxf(mx, acc[k]); }
  float sum = 0.f;
#pragma unroll
  for (int k = 0; k < KK; k++) { acc[k] = expf(acc[k] - mx); sum += acc[k]; }
  float inv = 1.0f / sum;
#pragma unroll
  for (int k = 0; k < KK; k++)
    A[((size_t)(b * HH + h) * KK + k) * NN + n] = acc[k] * inv;
}

// ---- upd[b,k,h*64+d] = (sum_n attn*V) / (sum_n attn + 1e-8)  (normalizer fused in) ----
__global__ __launch_bounds__(64) void k_upd(const float* __restrict__ A,
                                            const float* __restrict__ V,
                                            float* __restrict__ U) {
  int blk = blockIdx.x;            // b*KK*HH + k*HH + h
  int h = blk % HH;
  int k = (blk / HH) % KK;
  int b = blk / (HH * KK);
  int d = threadIdx.x;
  const float* a = A + ((size_t)(b * HH + h) * KK + k) * NN;
  const float* v = V + (size_t)b * NN * DD + h * DH + d;
  float acc = 0.f, s = 0.f;
  for (int n = 0; n < NN; n++) {
    float w = a[n];
    s += w;
    acc += w * v[(size_t)n * DD];
  }
  U[((size_t)(b * KK + k)) * DD + h * DH + d] = acc / (s + 1e-8f);
}

// ---- GRU cell, one block per slot row ----
__global__ __launch_bounds__(256) void k_gru(const float* __restrict__ U,
                                             float* __restrict__ slots,
                                             const float* __restrict__ Wih,
                                             const float* __restrict__ Whh,
                                             const float* __restrict__ bih,
                                             const float* __restrict__ bhh) {
  __shared__ float x[256], hs[256];
  int row = blockIdx.x, t = threadIdx.x;
  x[t]  = U[(size_t)row * DD + t];
  hs[t] = slots[(size_t)row * DD + t];
  __syncthreads();
  float gir = bih[t], giz = bih[DD + t], gin = bih[2 * DD + t];
  float ghr = bhh[t], ghz = bhh[DD + t], ghn = bhh[2 * DD + t];
  for (int j = 0; j < DD; j++) {
    float xv = x[j], hv = hs[j];
    const float* wi = Wih + (size_t)j * 3 * DD;
    const float* wh = Whh + (size_t)j * 3 * DD;
    gir += xv * wi[t];      giz += xv * wi[DD + t];  gin += xv * wi[2 * DD + t];
    ghr += hv * wh[t];      ghz += hv * wh[DD + t];  ghn += hv * wh[2 * DD + t];
  }
  float r  = 1.0f / (1.0f + expf(-(gir + ghr)));
  float z  = 1.0f / (1.0f + expf(-(giz + ghz)));
  float nn = tanhf(gin + r * ghn);
  slots[(size_t)row * DD + t] = (1.0f - z) * nn + z * hs[t];
}

// ---- FFN stage 1: gelu(ln(slots) @ W1 + b1) -> HBUF ----
__global__ __launch_bounds__(256) void k_ffn1(const float* __restrict__ slots,
                                              const float* __restrict__ g,
                                              const float* __restrict__ b,
                                              const float* __restrict__ W1,
                                              const float* __restrict__ b1,
                                              float* __restrict__ Hb) {
  __shared__ float red[256], sl[256];
  int row = blockIdx.x, t = threadIdx.x;
  float v = slots[(size_t)row * DD + t];
  float s1 = block_reduce_sum(v, red);
  float s2 = block_reduce_sum(v * v, red);
  float m = s1 * (1.0f / DD);
  float var = s2 * (1.0f / DD) - m * m;
  float rs = rsqrtf(var + 1e-5f);
  sl[t] = (v - m) * rs * g[t] + b[t];
  __syncthreads();
  for (int q = 0; q < 4; q++) {
    int c = t + q * DD;
    float acc = b1[c];
    for (int j = 0; j < DD; j++) acc += sl[j] * W1[(size_t)j * 4 * DD + c];
    float u = 0.7978845608f * (acc + 0.044715f * acc * acc * acc);
    Hb[(size_t)row * 4 * DD + c] = 0.5f * acc * (1.0f + tanhf(u));
  }
}

// ---- FFN stage 2: slots += HBUF @ W2 + b2 ----
__global__ __launch_bounds__(256) void k_ffn2(const float* __restrict__ Hb,
                                              const float* __restrict__ W2,
                                              const float* __restrict__ b2,
                                              float* __restrict__ slots) {
  __shared__ float hl[1024];
  int row = blockIdx.x, t = threadIdx.x;
  for (int q = 0; q < 4; q++) hl[t + q * 256] = Hb[(size_t)row * 1024 + t + q * 256];
  __syncthreads();
  float acc = b2[t];
  for (int j = 0; j < 1024; j++) acc += hl[j] * W2[(size_t)j * DD + t];
  slots[(size_t)row * DD + t] += acc;
}

// ---- attn mean over heads -> amap[b,k,n] ----
__global__ void k_amap(const float* __restrict__ A, float* __restrict__ M) {
  int idx = blockIdx.x * 256 + threadIdx.x;
  if (idx >= BB * KK * NN) return;
  int n = idx % NN;
  int k = (idx / NN) % KK;
  int b = idx / (NN * KK);
  float s = 0.f;
  for (int h = 0; h < HH; h++) s += A[((size_t)(b * HH + h) * KK + k) * NN + n];
  M[idx] = s * 0.25f;
}

// ---- centroids ----
__global__ __launch_bounds__(256) void k_cent(const float* __restrict__ M, float* __restrict__ C) {
  __shared__ float red[256];
  int bk = blockIdx.x;
  const float* m = M + (size_t)bk * NN;
  float sw = 0.f, sx = 0.f, sy = 0.f;
  for (int n = threadIdx.x; n < NN; n += 256) {
    float w = m[n];
    sw += w;
    sx += w * ((float)(n % 28) * (1.0f / 27.0f));
    sy += w * ((float)(n / 28) * (1.0f / 27.0f));
  }
  sw = block_reduce_sum(sw, red);
  sx = block_reduce_sum(sx, red);
  sy = block_reduce_sum(sy, red);
  if (threadIdx.x == 0) {
    float inv = 1.0f / (sw + 1e-8f);
    C[bk * 2 + 0] = sx * inv;
    C[bk * 2 + 1] = sy * inv;
  }
}

// ---- hilbert code + stable argsort + gather ----
__global__ __launch_bounds__(256) void k_sort(const float* __restrict__ slots,
                                              const float* __restrict__ cent,
                                              float* __restrict__ oslots,
                                              float* __restrict__ ocent) {
  __shared__ int code[KK], ord[KK];
  int b = blockIdx.x, t = threadIdx.x;
  if (t < KK) {
    float cx = cent[(b * KK + t) * 2 + 0];
    float cy = cent[(b * KK + t) * 2 + 1];
    int x = (int)fminf(fmaxf(cx * 15.0f, 0.0f), 15.0f);
    int y = (int)fminf(fmaxf(cy * 15.0f, 0.0f), 15.0f);
    int d = 0;
    for (int s = 8; s > 0; s >>= 1) {
      int rx = (x & s) ? 1 : 0;
      int ry = (y & s) ? 1 : 0;
      d += s * s * ((3 * rx) ^ ry);
      int fx = rx ? (s - 1 - x) : x;
      int fy = rx ? (s - 1 - y) : y;
      int nx2 = (ry == 0) ? fy : x;
      int ny2 = (ry == 0) ? fx : y;
      x = nx2; y = ny2;
    }
    code[t] = d;
  }
  __syncthreads();
  if (t == 0) {
    for (int i = 0; i < KK; i++) ord[i] = i;
    for (int i = 1; i < KK; i++) {   // stable insertion sort (ties keep index order)
      int key = ord[i]; int kd = code[key]; int j = i - 1;
      while (j >= 0 && code[ord[j]] > kd) { ord[j + 1] = ord[j]; j--; }
      ord[j + 1] = key;
    }
  }
  __syncthreads();
  for (int i = t; i < KK * DD; i += 256) {
    int k = i / DD, d = i % DD;
    oslots[((size_t)(b * KK + k)) * DD + d] = slots[((size_t)(b * KK + ord[k])) * DD + d];
  }
  if (t < KK * 2) {
    int k = t / 2, c = t % 2;
    ocent[(b * KK + k) * 2 + c] = cent[(b * KK + ord[k]) * 2 + c];
  }
}

// ---- FGW + Sinkhorn, one block (64 threads) per batch, all in LDS ----
__global__ __launch_bounds__(64) void k_fgw(const float* __restrict__ sq, const float* __restrict__ sr,
                                            const float* __restrict__ cq, const float* __restrict__ cr,
                                            const float* __restrict__ W1, const float* __restrict__ b1,
                                            const float* __restrict__ W2, const float* __restrict__ b2,
                                            const float* __restrict__ log_eps, float* __restrict__ out) {
  __shared__ float srow[DD], h1[DD];
  __shared__ float pq[KK][DD], pr[KK][DD];
  __shared__ float Cm[64], Sq[64], Sr[64], Tm[64], Cf[64], lgK[64], TS[64], red[64];
  __shared__ float lu[KK], lv[KK], rsum[KK], csum[KK], t1[KK], t2[KK];
  int b = blockIdx.x, t = threadIdx.x;
  // proj(s) = relu(s@W1+b1)@W2+b2 for both sides
  for (int side = 0; side < 2; side++) {
    const float* S = side ? sr : sq;
    for (int k = 0; k < KK; k++) {
      for (int j = t; j < DD; j += 64) srow[j] = S[((size_t)(b * KK + k)) * DD + j];
      __syncthreads();
      for (int q = 0; q < 4; q++) {
        int c = t + q * 64;
        float acc = b1[c];
        for (int j = 0; j < DD; j++) acc += srow[j] * W1[(size_t)j * DD + c];
        h1[c] = fmaxf(acc, 0.0f);
      }
      __syncthreads();
      for (int q = 0; q < 4; q++) {
        int c = t + q * 64;
        float acc = b2[c];
        for (int j = 0; j < DD; j++) acc += h1[j] * W2[(size_t)j * DD + c];
        if (side) pr[k][c] = acc; else pq[k][c] = acc;
      }
      __syncthreads();
    }
  }
  int i = t / KK, j = t % KK;
  {
    float s = 0.f;
    for (int d = 0; d < DD; d++) { float df = pq[i][d] - pr[j][d]; s += df * df; }
    Cm[t] = sqrtf(fmaxf(s, 1e-6f));
    float dxq = cq[(b * KK + i) * 2] - cq[(b * KK + j) * 2];
    float dyq = cq[(b * KK + i) * 2 + 1] - cq[(b * KK + j) * 2 + 1];
    Sq[t] = sqrtf(fmaxf(dxq * dxq + dyq * dyq, 1e-6f));
    float dxr = cr[(b * KK + i) * 2] - cr[(b * KK + j) * 2];
    float dyr = cr[(b * KK + i) * 2 + 1] - cr[(b * KK + j) * 2 + 1];
    Sr[t] = sqrtf(fmaxf(dxr * dxr + dyr * dyr, 1e-6f));
    Tm[t] = 1.0f / 64.0f;
  }
  __syncthreads();
  float eps = expf(log_eps[0]);
  eps = fminf(fmaxf(eps, 0.01f), 0.5f);
  float rho = 0.1f / (0.1f + eps);
  float lmu = logf(1.0f / KK);
  for (int outer = 0; outer < 5; outer++) {
    if (t < KK) {
      float rs_ = 0.f, cs_ = 0.f;
      for (int m = 0; m < KK; m++) { rs_ += Tm[t * KK + m]; cs_ += Tm[m * KK + t]; }
      rsum[t] = rs_; csum[t] = cs_;
    }
    __syncthreads();
    if (t < KK) {
      float a = 0.f, c2 = 0.f;
      for (int m = 0; m < KK; m++) {
        float q2 = Sq[t * KK + m]; a  += q2 * q2 * rsum[m];
        float r2 = Sr[m * KK + t]; c2 += csum[m] * r2 * r2;
      }
      t1[t] = a; t2[t] = c2;
    }
    float tsv = 0.f;
    for (int m = 0; m < KK; m++) tsv += Tm[i * KK + m] * Sr[m * KK + j];
    TS[t] = tsv;
    __syncthreads();
    {
      float s = 0.f;
      for (int m = 0; m < KK; m++) s += Sq[i * KK + m] * TS[m * KK + j];
      float L = t1[i] + t2[j] - 2.0f * s;
      float cf = 0.5f * Cm[t] + 0.5f * L;
      Cf[t] = cf; lgK[t] = -cf / eps;
    }
    if (t < KK) { lu[t] = 0.f; lv[t] = 0.f; }
    __syncthreads();
    for (int it = 0; it < 10; it++) {
      if (t < KK) {
        float mx = -1e30f;
        for (int m = 0; m < KK; m++) mx = fmaxf(mx, lgK[t * KK + m] + lv[m]);
        float s = 0.f;
        for (int m = 0; m < KK; m++) s += expf(lgK[t * KK + m] + lv[m] - mx);
        lu[t] = rho * (lmu - (mx + logf(s)));
      }
      __syncthreads();
      if (t < KK) {
        float mx = -1e30f;
        for (int m = 0; m < KK; m++) mx = fmaxf(mx, lgK[m * KK + t] + lu[m]);
        float s = 0.f;
        for (int m = 0; m < KK; m++) s += expf(lgK[m * KK + t] + lu[m] - mx);
        lv[t] = rho * (lmu - (mx + logf(s)));
      }
      __syncthreads();
    }
    Tm[t] = expf(lu[i] + lgK[t] + lv[j]);
    __syncthreads();
  }
  red[t] = Tm[t] * Cf[t];
  __syncthreads();
  for (int s = 32; s > 0; s >>= 1) { if (t < s) red[t] += red[t + s]; __syncthreads(); }
  if (t == 0) out[b] = 1.0f / (1.0f + expf(red[0]));   // sigmoid(-cost)
}

extern "C" void kernel_launch(void* const* d_in, const int* in_sizes, int n_in,
                              void* d_out, int out_size, void* d_ws, size_t ws_size,
                              hipStream_t stream) {
  (void)in_sizes; (void)out_size; (void)ws_size;
  if (n_in < 25) return;
  const float* feat_q     = (const float*)d_in[0];
  const float* feat_r     = (const float*)d_in[1];
  const float* W_q        = (const float*)d_in[2];
  const float* W_k        = (const float*)d_in[3];
  const float* W_v        = (const float*)d_in[4];
  const float* gru_Wih    = (const float*)d_in[5];
  const float* gru_Whh    = (const float*)d_in[6];
  const float* gru_bih    = (const float*)d_in[7];
  const float* gru_bhh    = (const float*)d_in[8];
  const float* ln_in_g    = (const float*)d_in[9];
  const float* ln_in_b    = (const float*)d_in[10];
  const float* ln_s_g     = (const float*)d_in[11];
  const float* ln_s_b     = (const float*)d_in[12];
  const float* ff_ln_g    = (const float*)d_in[13];
  const float* ff_ln_b    = (const float*)d_in[14];
  const float* ff_W1      = (const float*)d_in[15];
  const float* ff_b1      = (const float*)d_in[16];
  const float* ff_W2      = (const float*)d_in[17];
  const float* ff_b2      = (const float*)d_in[18];
  const float* cp_W1      = (const float*)d_in[19];
  const float* cp_b1      = (const float*)d_in[20];
  const float* cp_W2      = (const float*)d_in[21];
  const float* cp_b2      = (const float*)d_in[22];
  const float* log_eps    = (const float*)d_in[23];
  const float* slots_init = (const float*)d_in[24];
  float* out = (float*)d_out;

  char* ws = (char*)d_ws;
  size_t off = 0;
  auto alloc = [&](size_t bytes) -> size_t {
    size_t o = off; off += (bytes + 255) & ~(size_t)255; return o;
  };
  __bf16* XLN = (__bf16*)(ws + alloc((size_t)BB * NN * FF * 2));
  __bf16* WKT = (__bf16*)(ws + alloc((size_t)DD * FF * 2));
  __bf16* WVT = (__bf16*)(ws + alloc((size_t)DD * FF * 2));
  float* KM   = (float*)(ws + alloc((size_t)BB * NN * DD * 4));
  float* VM   = (float*)(ws + alloc((size_t)BB * NN * DD * 4));
  float* SLOTS[2];
  SLOTS[0] = (float*)(ws + alloc((size_t)BB * KK * DD * 4));
  SLOTS[1] = (float*)(ws + alloc((size_t)BB * KK * DD * 4));
  float* QM   = (float*)(ws + alloc((size_t)BB * KK * DD * 4));
  float* ATTN = (float*)(ws + alloc((size_t)BB * HH * KK * NN * 4));
  float* UPD  = (float*)(ws + alloc((size_t)BB * KK * DD * 4));
  float* HBUF = (float*)(ws + alloc((size_t)BB * KK * 4 * DD * 4));
  float* AMAP = (float*)(ws + alloc((size_t)BB * KK * NN * 4));
  float* CENT[2];
  CENT[0] = (float*)(ws + alloc((size_t)BB * KK * 2 * 4));
  CENT[1] = (float*)(ws + alloc((size_t)BB * KK * 2 * 4));
  float* SSORT[2];
  SSORT[0] = (float*)(ws + alloc((size_t)BB * KK * DD * 4));
  SSORT[1] = (float*)(ws + alloc((size_t)BB * KK * DD * 4));
  float* CSORT[2];
  CSORT[0] = (float*)(ws + alloc((size_t)BB * KK * 2 * 4));
  CSORT[1] = (float*)(ws + alloc((size_t)BB * KK * 2 * 4));

  k_wt<<<(FF * DD + 255) / 256, 256, 0, stream>>>(W_k, WKT);
  k_wt<<<(FF * DD + 255) / 256, 256, 0, stream>>>(W_v, WVT);

  for (int side = 0; side < 2; side++) {
    const float* feats = side ? feat_r : feat_q;
    k_ln_feat<<<BB * NN, 256, 0, stream>>>(feats, ln_in_g, ln_in_b, XLN);
    k_gemm<<<BB * NN / 32, 256, 0, stream>>>(XLN, WKT, WVT, KM, VM);
    k_slot_init<<<(BB * KK * DD + 255) / 256, 256, 0, stream>>>(slots_init, SLOTS[side]);
    for (int it = 0; it < 3; it++) {
      k_qproj<<<BB * KK, 256, 0, stream>>>(SLOTS[side], ln_s_g, ln_s_b, W_q, QM);
      k_attn<<<(BB * HH * NN + 127) / 128, 128, 0, stream>>>(QM, KM, ATTN);
      k_upd<<<BB * KK * HH, 64, 0, stream>>>(ATTN, VM, UPD);
      k_gru<<<BB * KK, 256, 0, stream>>>(UPD, SLOTS[side], gru_Wih, gru_Whh, gru_bih, gru_bhh);
      k_ffn1<<<BB * KK, 256, 0, stream>>>(SLOTS[side], ff_ln_g, ff_ln_b, ff_W1, ff_b1, HBUF);
      k_ffn2<<<BB * KK, 256, 0, stream>>>(HBUF, ff_W2, ff_b2, SLOTS[side]);
    }
    k_amap<<<(BB * KK * NN + 255) / 256, 256, 0, stream>>>(ATTN, AMAP);
    k_cent<<<BB * KK, 256, 0, stream>>>(AMAP, CENT[side]);
    k_sort<<<BB, 256, 0, stream>>>(SLOTS[side], CENT[side], SSORT[side], CSORT[side]);
  }
  k_fgw<<<BB, 64, 0, stream>>>(SSORT[0], SSORT[1], CSORT[0], CSORT[1],
                               cp_W1, cp_b1, cp_W2, cp_b2, log_eps, out);
}
